// GNN_Node_Classifier_12472585027649
// MI455X (gfx1250) — compile-verified, matrix operand order checked by reading, not run on hardware
//
#include <hip/hip_runtime.h>
#include <hip/hip_bf16.h>

// ---------------------------------------------------------------------------
// GNN node classifier for MI455X (gfx1250, wave32):
//   buf0 = X @ W1                               (bf16x3-split WMMA GEMM)
//   buf1 = segment_sum(buf0[src]*ew -> dst)     (f32 atomics, fits in 192MB L2)
//   buf0 = relu(buf1 + b1) @ W2                 (relu+bias fused in A-load)
//   buf1 = segment_sum(buf0[src]*ew -> dst)
//   pooled = sum_n relu(buf1 + b2)              (fused, H2 never materialized)
//   out = softmax(pooled @ wd + bd)
// ---------------------------------------------------------------------------

typedef __attribute__((ext_vector_type(16))) __bf16 v16bf;
typedef __attribute__((ext_vector_type(8)))  __bf16 v8bf;
typedef __attribute__((ext_vector_type(8)))  float  v8f;

#define N_NODES   50000
#define N_EDGES   800000
#define N_FEAT    1024
#define N_HID     512
#define N_CLASSES 4

__device__ __forceinline__ unsigned short f2bf_rne(float f) {
  unsigned u = __float_as_uint(f);
  unsigned r = u + 0x7FFFu + ((u >> 16) & 1u);
  return (unsigned short)(r >> 16);
}
__device__ __forceinline__ float bf2f(unsigned short h) {
  return __uint_as_float(((unsigned)h) << 16);
}
__device__ __forceinline__ uint2 pack4bf(const unsigned short h[4]) {
  uint2 v;
  v.x = (unsigned)h[0] | ((unsigned)h[1] << 16);
  v.y = (unsigned)h[2] | ((unsigned)h[3] << 16);
  return v;
}

#define BM 128
#define BN 64
#define BK 32
#define ST 40   // LDS row stride in bf16 elements (BK + 8 pad -> bank spread)

// C[M,N] = A[M,K] @ B[K,N], fp32 in/out, internally bf16 hi/lo 3-product split.
// HAS_BIAS: A is replaced by relu(A + bias[k]) on load (GCN layer-2 fusion).
template <bool HAS_BIAS>
__global__ __launch_bounds__(256) void gemm_bf16x3_wmma(
    const float* __restrict__ A, const float* __restrict__ B,
    float* __restrict__ C, int M, int K, int N,
    const float* __restrict__ bias) {
  __shared__ unsigned short Ahi[BM * ST];
  __shared__ unsigned short Alo[BM * ST];
  __shared__ unsigned short Bhi[BN * ST];   // stored transposed: [n][k]
  __shared__ unsigned short Blo[BN * ST];

  const int tid   = threadIdx.x;
  const int lane  = tid & 31;
  const int wid   = tid >> 5;
  const int waveM = wid >> 1;     // 0..3
  const int waveN = wid & 1;      // 0..1
  const int m0 = blockIdx.x * BM;
  const int n0 = blockIdx.y * BN;

  v8f acc[2][2];
#pragma unroll
  for (int mi = 0; mi < 2; ++mi)
#pragma unroll
    for (int ni = 0; ni < 2; ++ni)
#pragma unroll
      for (int e = 0; e < 8; ++e) acc[mi][ni][e] = 0.f;

  const int aCol4 = (tid & 7) * 4;    // k offset within A tile (0..28)
  const int aRow  = tid >> 3;         // 0..31
  const int bN    = tid & 63;         // n column within B tile (0..63)
  const int bK4   = (tid >> 6) * 4;   // k base within B tile (0,4,8,12)

  for (int k0 = 0; k0 < K; k0 += BK) {
    __syncthreads();

    // Bias for this thread's 4 A-columns: one float4 per K-step (uniform over rows).
    float bias4[4] = {0.f, 0.f, 0.f, 0.f};
    if (HAS_BIAS) {
      float4 bv = *reinterpret_cast<const float4*>(bias + k0 + aCol4);
      bias4[0] = bv.x; bias4[1] = bv.y; bias4[2] = bv.z; bias4[3] = bv.w;
    }

    // ---- stage A tile: BM x BK f32 -> hi/lo bf16 in LDS (packed b64 stores) ----
#pragma unroll
    for (int i = 0; i < 4; ++i) {
      int rr = aRow + 32 * i;
      int gr = m0 + rr;
      float4 av = make_float4(0.f, 0.f, 0.f, 0.f);
      if (gr < M) {
        av = *reinterpret_cast<const float4*>(A + (size_t)gr * K + k0 + aCol4);
        if (k0 + BK < K)  // speculative prefetch of next K tile
          __builtin_prefetch(A + (size_t)gr * K + k0 + BK + aCol4, 0, 1);
      }
      float vv[4] = {av.x, av.y, av.z, av.w};
      unsigned short hh[4], ll[4];
#pragma unroll
      for (int j = 0; j < 4; ++j) {
        float f = vv[j];
        if (HAS_BIAS) f = fmaxf(f + bias4[j], 0.f);
        hh[j] = f2bf_rne(f);
        ll[j] = f2bf_rne(f - bf2f(hh[j]));
      }
      *reinterpret_cast<uint2*>(&Ahi[rr * ST + aCol4]) = pack4bf(hh);
      *reinterpret_cast<uint2*>(&Alo[rr * ST + aCol4]) = pack4bf(ll);
    }

    // ---- stage B tile: BK x BN f32 -> transposed hi/lo bf16 in LDS ----
    // Each thread owns column n and 4 consecutive k: loads are coalesced across
    // the 64 consecutive-n lanes; transposed store becomes a packed b64.
#pragma unroll
    for (int i = 0; i < 2; ++i) {
      int kb = bK4 + 16 * i;  // 0..28
      unsigned short hh[4], ll[4];
#pragma unroll
      for (int j = 0; j < 4; ++j) {
        float f = B[(size_t)(k0 + kb + j) * N + n0 + bN];
        hh[j] = f2bf_rne(f);
        ll[j] = f2bf_rne(f - bf2f(hh[j]));
      }
      *reinterpret_cast<uint2*>(&Bhi[bN * ST + kb]) = pack4bf(hh);
      *reinterpret_cast<uint2*>(&Blo[bN * ST + kb]) = pack4bf(ll);
    }
    __syncthreads();

    // ---- fragments (ISA 7.12.2: lane<16 -> K{0..7,16..23}; lane>=16 -> K{8..15,24..31}) ----
    const int K0 = (lane < 16) ? 0 : 8;
    v16bf fahi[2], falo[2], fbhi[2], fblo[2];
#pragma unroll
    for (int mi = 0; mi < 2; ++mi) {
      int r = waveM * 32 + mi * 16 + (lane & 15);
      v8bf h0 = *reinterpret_cast<const v8bf*>(&Ahi[r * ST + K0]);
      v8bf h1 = *reinterpret_cast<const v8bf*>(&Ahi[r * ST + K0 + 16]);
      v8bf l0 = *reinterpret_cast<const v8bf*>(&Alo[r * ST + K0]);
      v8bf l1 = *reinterpret_cast<const v8bf*>(&Alo[r * ST + K0 + 16]);
#pragma unroll
      for (int e = 0; e < 8; ++e) {
        fahi[mi][e] = h0[e]; fahi[mi][e + 8] = h1[e];
        falo[mi][e] = l0[e]; falo[mi][e + 8] = l1[e];
      }
    }
#pragma unroll
    for (int ni = 0; ni < 2; ++ni) {
      int c = waveN * 32 + ni * 16 + (lane & 15);
      v8bf h0 = *reinterpret_cast<const v8bf*>(&Bhi[c * ST + K0]);
      v8bf h1 = *reinterpret_cast<const v8bf*>(&Bhi[c * ST + K0 + 16]);
      v8bf l0 = *reinterpret_cast<const v8bf*>(&Blo[c * ST + K0]);
      v8bf l1 = *reinterpret_cast<const v8bf*>(&Blo[c * ST + K0 + 16]);
#pragma unroll
      for (int e = 0; e < 8; ++e) {
        fbhi[ni][e] = h0[e]; fbhi[ni][e + 8] = h1[e];
        fblo[ni][e] = l0[e]; fblo[ni][e + 8] = l1[e];
      }
    }

    // ---- 3-product split-precision accumulate: hi*hi + hi*lo + lo*hi ----
#pragma unroll
    for (int mi = 0; mi < 2; ++mi)
#pragma unroll
      for (int ni = 0; ni < 2; ++ni) {
        acc[mi][ni] = __builtin_amdgcn_wmma_f32_16x16x32_bf16(
            false, fahi[mi], false, fbhi[ni], (short)0, acc[mi][ni], false, false);
        acc[mi][ni] = __builtin_amdgcn_wmma_f32_16x16x32_bf16(
            false, fahi[mi], false, fblo[ni], (short)0, acc[mi][ni], false, false);
        acc[mi][ni] = __builtin_amdgcn_wmma_f32_16x16x32_bf16(
            false, falo[mi], false, fbhi[ni], (short)0, acc[mi][ni], false, false);
      }
  }

  // ---- store C (8-VGPR f32 C/D layout: elem r -> M = r + 8*(lane>=16), N = lane%16) ----
  const int rowHi = ((lane >> 4) & 1) * 8;
#pragma unroll
  for (int mi = 0; mi < 2; ++mi)
#pragma unroll
    for (int ni = 0; ni < 2; ++ni) {
      int gc = n0 + waveN * 32 + ni * 16 + (lane & 15);
#pragma unroll
      for (int r = 0; r < 8; ++r) {
        int gr = m0 + waveM * 32 + mi * 16 + r + rowHi;
        if (gr < M) C[(size_t)gr * N + gc] = acc[mi][ni][r];
      }
    }
}

__global__ void zero_kernel(float* __restrict__ p, size_t n) {
  size_t i = (size_t)blockIdx.x * blockDim.x + threadIdx.x;
  size_t stride = (size_t)gridDim.x * blockDim.x;
  for (; i < n; i += stride) p[i] = 0.f;
}

// agg[dst[e]] += H[src[e]] * ew[e]   (features fastest: coalesced float4 gather;
// agg is 102.4MB -> resident in 192MB L2, atomics execute at L2 rate)
__global__ __launch_bounds__(256) void scatter_kernel(
    const float* __restrict__ H, const int* __restrict__ src,
    const int* __restrict__ dst, const float* __restrict__ ew,
    float* __restrict__ agg, int nEdges) {
  unsigned gid = blockIdx.x * blockDim.x + threadIdx.x;
  unsigned f4 = gid & 127;        // 128 float4 chunks = 512 feats
  unsigned e  = gid >> 7;
  if (e >= (unsigned)nEdges) return;
  int s = src[e], d = dst[e];
  float w = ew[e];
  float4 h = reinterpret_cast<const float4*>(H + (size_t)s * N_HID)[f4];
  float* o = agg + (size_t)d * N_HID + (size_t)f4 * 4;
  atomicAdd(o + 0, h.x * w);
  atomicAdd(o + 1, h.y * w);
  atomicAdd(o + 2, h.z * w);
  atomicAdd(o + 3, h.w * w);
}

// pooled[f] += sum_{n in chunk} relu(agg[n][f] + b[f])
__global__ __launch_bounds__(128) void pool_kernel(
    const float* __restrict__ agg, const float* __restrict__ b,
    float* __restrict__ pooled, int nNodes) {
  int f = blockIdx.y * 128 + threadIdx.x;
  int nStart = blockIdx.x * 512;
  int nEnd = nStart + 512; if (nEnd > nNodes) nEnd = nNodes;
  float bias = b[f];
  float s = 0.f;
  for (int n = nStart; n < nEnd; ++n)
    s += fmaxf(agg[(size_t)n * N_HID + f] + bias, 0.f);
  atomicAdd(&pooled[f], s);
}

__global__ void head_kernel(const float* __restrict__ pooled,
                            const float* __restrict__ wd,
                            const float* __restrict__ bd,
                            float* __restrict__ out) {
  if (threadIdx.x != 0 || blockIdx.x != 0) return;
  float logits[N_CLASSES];
#pragma unroll
  for (int c = 0; c < N_CLASSES; ++c) {
    float s = bd[c];
    for (int f = 0; f < N_HID; ++f) s += pooled[f] * wd[f * N_CLASSES + c];
    logits[c] = s;
  }
  float m = logits[0];
#pragma unroll
  for (int c = 1; c < N_CLASSES; ++c) m = fmaxf(m, logits[c]);
  float ex[N_CLASSES], sum = 0.f;
#pragma unroll
  for (int c = 0; c < N_CLASSES; ++c) { ex[c] = __expf(logits[c] - m); sum += ex[c]; }
#pragma unroll
  for (int c = 0; c < N_CLASSES; ++c) out[c] = ex[c] / sum;
}

extern "C" void kernel_launch(void* const* d_in, const int* in_sizes, int n_in,
                              void* d_out, int out_size, void* d_ws, size_t ws_size,
                              hipStream_t stream) {
  const float* x    = (const float*)d_in[0];
  const int*   esrc = (const int*)d_in[1];
  const int*   edst = (const int*)d_in[2];
  const float* ew   = (const float*)d_in[3];
  const float* w1   = (const float*)d_in[4];
  const float* b1   = (const float*)d_in[5];
  const float* w2   = (const float*)d_in[6];
  const float* b2   = (const float*)d_in[7];
  const float* wd   = (const float*)d_in[8];
  const float* bd   = (const float*)d_in[9];
  float* out = (float*)d_out;

  const size_t HN = (size_t)N_NODES * N_HID;     // 25.6M floats
  float* buf0   = (float*)d_ws;                  // H pre-aggregation
  float* buf1   = buf0 + HN;                     // aggregation target
  float* pooled = buf1 + HN;                     // 512 floats

  dim3 gGemm((N_NODES + BM - 1) / BM, N_HID / BN);
  const int scatterBlocks = (N_EDGES * 128) / 256;  // 400000

  // Layer 1: buf0 = X @ W1
  gemm_bf16x3_wmma<false><<<gGemm, 256, 0, stream>>>(x, w1, buf0, N_NODES, N_FEAT, N_HID, nullptr);
  zero_kernel<<<2048, 256, 0, stream>>>(buf1, HN);
  scatter_kernel<<<scatterBlocks, 256, 0, stream>>>(buf0, esrc, edst, ew, buf1, N_EDGES);

  // Layer 2: buf0 = relu(buf1 + b1) @ W2   (relu+bias fused into GEMM A-load)
  gemm_bf16x3_wmma<true><<<gGemm, 256, 0, stream>>>(buf1, w2, buf0, N_NODES, N_HID, N_HID, b1);
  zero_kernel<<<2048, 256, 0, stream>>>(buf1, HN);
  scatter_kernel<<<scatterBlocks, 256, 0, stream>>>(buf0, esrc, edst, ew, buf1, N_EDGES);

  // Pool (relu+b2 fused) + head + softmax
  zero_kernel<<<1, 256, 0, stream>>>(pooled, (size_t)N_HID);
  pool_kernel<<<dim3((N_NODES + 511) / 512, N_HID / 128), 128, 0, stream>>>(buf1, b2, pooled, N_NODES);
  head_kernel<<<1, 32, 0, stream>>>(pooled, wd, bd, out);
}
